// CPCNet_33543694581850
// MI455X (gfx1250) — matrix operand order repaired and checked by reading, not compiled, not run on hardware
//
#include <hip/hip_runtime.h>

typedef __attribute__((ext_vector_type(16))) _Float16 v16h;
typedef __attribute__((ext_vector_type(8)))  _Float16 v8h;
typedef __attribute__((ext_vector_type(8)))  float    v8f;
typedef __attribute__((ext_vector_type(4)))  float    v4f;

#define BATCH 64
#define L_IN 512
#define T1 255
#define TS 254
#define ZD 256
#define NNEG 15
#define OFF0 16
#define STEPS 12
#define RES_SCALE 0.70710678118654752f

// -------- encoder conv1: x[B,3,512] -> h1[B,255,32], k=4 s=2, relu --------
__global__ __launch_bounds__(256) void conv1_kernel(const float* __restrict__ x,
    const float* __restrict__ w, const float* __restrict__ bias,
    float* __restrict__ h1) {
  int id = blockIdx.x * 256 + threadIdx.x;
  if (id >= BATCH * T1 * 32) return;
  int co = id & 31;
  int t  = (id >> 5) % T1;
  int b  = id / (32 * T1);
  float acc = bias[co];
#pragma unroll
  for (int ci = 0; ci < 3; ci++)
#pragma unroll
    for (int j = 0; j < 4; j++)
      acc += x[(b * 3 + ci) * L_IN + 2 * t + j] * w[(co * 3 + ci) * 4 + j];
  h1[id] = fmaxf(acc, 0.f);
}

// -------- encoder conv2: h1[B,255,32] -> h2[B,254,64], k=2 s=1, relu --------
__global__ __launch_bounds__(256) void conv2_kernel(const float* __restrict__ h1,
    const float* __restrict__ w, const float* __restrict__ bias,
    float* __restrict__ h2) {
  int id = blockIdx.x * 256 + threadIdx.x;
  if (id >= BATCH * TS * 64) return;
  int co = id & 63;
  int t  = (id >> 6) % TS;
  int b  = id / (64 * TS);
  float acc = bias[co];
#pragma unroll
  for (int j = 0; j < 2; j++) {
    const float* hrow = h1 + ((size_t)(b * T1 + t + j)) * 32;
#pragma unroll
    for (int ci = 0; ci < 32; ci++)
      acc += hrow[ci] * w[(co * 32 + ci) * 2 + j];
  }
  h2[id] = fmaxf(acc, 0.f);
}

// -------- weight prep: in[O][I][Kk] fp32 -> WMMA-fragment pack f16 ----------
// Pack layout: [j][ci/32][o/16][lane][half], lane/half in this code's contiguous-K order:
//   B 32x16 f16: lane l, half h -> N = l%16, K = h + 16*(l/16)
__global__ void wprep_kernel(const float* __restrict__ in, _Float16* __restrict__ out,
                             int O, int I, int Kk) {
  int id = blockIdx.x * 256 + threadIdx.x;
  int total = O * I * Kk;
  if (id >= total) return;
  int o   = id / (I * Kk);
  int rem = id - o * I * Kk;
  int ci  = rem / Kk;
  int j   = rem - ci * Kk;
  int kc   = ci >> 5;
  int kin  = ci & 31;
  int nt   = o >> 4;
  int lt   = (o & 15) + ((kin >> 4) << 4);
  int half = kin & 15;
  int Kc = I >> 5, Nt = O >> 4;
  size_t dst = ((((size_t)j * Kc + kc) * Nt + nt) * 32 + lt) * 16 + half;
  out[dst] = (_Float16)in[id];
}

// -------- pred prep: pred_w[c][o][k] fp32 -> per-step fragment pack f16 -----
__global__ void predprep_kernel(const float* __restrict__ in, _Float16* __restrict__ out) {
  int id = blockIdx.x * 256 + threadIdx.x;
  if (id >= ZD * ZD * STEPS) return;
  int c = id / (ZD * STEPS);
  int o = (id / STEPS) % ZD;
  int k = id % STEPS;
  int kc   = c >> 5;
  int kin  = c & 31;
  int nt   = o >> 4;
  int lt   = (o & 15) + ((kin >> 4) << 4);
  int half = kin & 15;
  size_t dst = ((((size_t)k * 8 + kc) * 16 + nt) * 32 + lt) * 16 + half;
  out[dst] = (_Float16)in[id];
}

// -------- generic causal-conv GEMM with WMMA, double-buffered LDS -----------
// C[M][N] = act?( sum_j A_shifted(j)[M][K] x Bpack[j][K][N] + bias )
// Block tile 128x128: 8 waves, each 16 rows x 8 WMMA n-tiles.
// A fp32 row-major [M][K], rows grouped T per batch; shift s=nshift-1-j
// gives causal left zero pad. B pre-packed in fragment order (see wprep).
// Fragment layouts used here (wave32):
//   A 16x32 f16: lane l, half h -> M=l%16, K = h + 16*(l/16)
//   C/D 16x16 f32: vgpr j -> M = j + 8*(l/16), N = l%16
__global__ __launch_bounds__(256) void gemm_wmma_kernel(
    const float* __restrict__ A, const _Float16* __restrict__ Bp,
    const float* __restrict__ bias, float* __restrict__ C,
    int N, int K, int T, int nshift, int relu) {
  const int Kc = K >> 5;           // K-chunks per shift
  const int Nt = N >> 4;           // total 16-col tiles in B pack
  const int nch = nshift * Kc;
  __shared__ __align__(32) v8h lA[2][8][32][2];   // 2 x 8KB
  __shared__ __align__(32) v8h lB[2][8][32][2];   // 2 x 8KB
  const int tid  = threadIdx.x;
  const int lane = tid & 31;
  const int wave = tid >> 5;
  const int rowBase  = blockIdx.y * 128;
  const int colTile0 = blockIdx.x * 8;

  v8f acc[8] = {};

  auto stage = [&](int c, int buf) {
    const int jj = c / Kc;
    const int kc = (c - jj * Kc) << 5;
    const int s  = nshift - 1 - jj;
    // ---- A tile (128 rows x 32 K) -> fragment order, f32->f16 ----
#pragma unroll
    for (int i = 0; i < 2; i++) {
      int q   = tid + (i << 8);          // 512 tasks
      int mt  = q >> 6;                  // m-tile 0..7
      int rem = q & 63;
      int l   = rem >> 1;                // lane slot
      int g   = rem & 1;                 // k-group
      int r   = rowBase + mt * 16 + (l & 15);
      int t   = r % T;
      int kk0 = kc + (g << 3) + ((l >> 4) << 4);
      v4f f0 = {0.f, 0.f, 0.f, 0.f}, f1 = {0.f, 0.f, 0.f, 0.f};
      if (t >= s) {
        const float* ap = A + (size_t)(r - s) * K + kk0;
        f0 = *(const v4f*)ap;
        f1 = *(const v4f*)(ap + 4);
      }
      v8h hv;
      hv[0] = (_Float16)f0.x; hv[1] = (_Float16)f0.y;
      hv[2] = (_Float16)f0.z; hv[3] = (_Float16)f0.w;
      hv[4] = (_Float16)f1.x; hv[5] = (_Float16)f1.y;
      hv[6] = (_Float16)f1.z; hv[7] = (_Float16)f1.w;
      lA[buf][mt][l][g] = hv;
    }
    // ---- B tile: already fragment-ordered in global, straight b128 copy ----
    const v8h* bbase = (const v8h*)(Bp +
        (((size_t)jj * Kc + (kc >> 5)) * Nt + colTile0) * 512);
#pragma unroll
    for (int i = 0; i < 2; i++) {
      int q   = tid + (i << 8);          // 512 tasks: 8 nt x 32 lanes x 2 grp
      int nt  = q >> 6;
      int rem = q & 63;
      int l   = rem >> 1;
      int g   = rem & 1;
      lB[buf][nt][l][g] = bbase[(nt * 32 + l) * 2 + g];
    }
  };

  stage(0, 0);
  __syncthreads();
  int buf = 0;
  for (int c = 0; c < nch; c++) {
    if (c + 1 < nch) stage(c + 1, buf ^ 1);   // overlap next-chunk staging
    v16h af = *(const v16h*)&lA[buf][wave][lane][0];
#pragma unroll
    for (int nt = 0; nt < 8; nt++) {
      v16h bf = *(const v16h*)&lB[buf][nt][lane][0];
      acc[nt] = __builtin_amdgcn_wmma_f32_16x16x32_f16(
          false, af, false, bf, (short)0, acc[nt], false, false);
    }
    __syncthreads();
    buf ^= 1;
  }

  // ---- epilogue: bias + optional relu, fp32 store ----
  const int mbase = rowBase + wave * 16 + ((lane >> 4) << 3);
  const int nloc  = lane & 15;
#pragma unroll
  for (int nt = 0; nt < 8; nt++) {
    int col = (colTile0 + nt) * 16 + nloc;
    float bv = bias ? bias[col] : 0.f;
#pragma unroll
    for (int jj = 0; jj < 8; jj++) {
      float v = acc[nt][jj] + bv;
      if (relu) v = fmaxf(v, 0.f);
      C[(size_t)(mbase + jj) * N + col] = v;
    }
  }
}

// -------- GroupNorm(1,C) + relu + scaled residual, one block per sample ----
__global__ __launch_bounds__(256) void gn_res_kernel(const float* __restrict__ conv,
    const float* __restrict__ resid, const float* __restrict__ gamma,
    const float* __restrict__ beta, float* __restrict__ out) {
  __shared__ float sb[256];
  __shared__ float sb2[256];
  const int b = blockIdx.x, tid = threadIdx.x;
  const size_t base = (size_t)b * TS * ZD;
  float s = 0.f, s2 = 0.f;
  for (int i = tid; i < TS * ZD; i += 256) {
    float v = conv[base + i];
    s += v; s2 += v * v;
  }
  sb[tid] = s; sb2[tid] = s2;
  __syncthreads();
  for (int st = 128; st > 0; st >>= 1) {
    if (tid < st) { sb[tid] += sb[tid + st]; sb2[tid] += sb2[tid + st]; }
    __syncthreads();
  }
  const float inv = 1.0f / (float)(TS * ZD);
  const float mean = sb[0] * inv;
  const float var  = sb2[0] * inv - mean * mean;
  const float rstd = rsqrtf(var + 1e-5f);
  for (int i = tid; i < TS * ZD; i += 256) {
    int c = i & (ZD - 1);
    float v = (conv[base + i] - mean) * rstd * gamma[c] + beta[c];
    v = fmaxf(v, 0.f);
    out[base + i] = (v + resid[base + i]) * RES_SCALE;
  }
}

// -------- CPC logits for one prediction step (gather negatives on the fly) --
__global__ __launch_bounds__(256) void cpc_kernel(const float* __restrict__ zp,
    const float* __restrict__ z, const int* __restrict__ neg_idxs,
    float* __restrict__ outp, int step, int Tlen, long long obase) {
  int id = blockIdx.x * 256 + threadIdx.x;
  if (id >= Tlen * BATCH * 16) return;
  int n = id & 15;
  int b = (id >> 4) & 63;
  int t = id >> 10;
  int toff = t + OFF0 + step;
  int trow;
  if (n == 0) {
    trow = b * TS + toff;                       // positive target
  } else {
    int raw = neg_idxs[(size_t)b * (TS * NNEG) + toff * NNEG + (n - 1)];
    int thresh = toff + b * BATCH;              // reference quirk: b*bsz
    trow = raw + (raw >= thresh ? 1 : 0);
  }
  const v4f* zr = (const v4f*)(zp + (size_t)(b * TS + t) * ZD);
  const v4f* tr = (const v4f*)(z + (size_t)trow * ZD);
  float s = 0.f;
#pragma unroll 8
  for (int q = 0; q < ZD / 4; q++) {
    v4f a = zr[q], c = tr[q];
    s += a.x * c.x + a.y * c.y + a.z * c.z + a.w * c.w;
  }
  outp[(size_t)obase + (size_t)(t * BATCH + b) * 16 + n] = s;
}

__global__ void zerof_kernel(float* __restrict__ p, long long start, long long end) {
  long long id = start + (long long)blockIdx.x * 256 + threadIdx.x;
  if (id < end) p[id] = 0.f;
}

extern "C" void kernel_launch(void* const* d_in, const int* in_sizes, int n_in,
                              void* d_out, int out_size, void* d_ws, size_t ws_size,
                              hipStream_t stream) {
  (void)in_sizes; (void)n_in; (void)ws_size;
  const float* x   = (const float*)d_in[0];
  const int*   neg = (const int*)d_in[1];
  const float* enc_w[4]  = {(const float*)d_in[2],  (const float*)d_in[3],
                            (const float*)d_in[4],  (const float*)d_in[5]};
  const float* enc_b[4]  = {(const float*)d_in[6],  (const float*)d_in[7],
                            (const float*)d_in[8],  (const float*)d_in[9]};
  const float* agg_w[4]  = {(const float*)d_in[10], (const float*)d_in[11],
                            (const float*)d_in[12], (const float*)d_in[13]};
  const float* agg_b[4]  = {(const float*)d_in[14], (const float*)d_in[15],
                            (const float*)d_in[16], (const float*)d_in[17]};
  const float* agg_g[4]  = {(const float*)d_in[18], (const float*)d_in[19],
                            (const float*)d_in[20], (const float*)d_in[21]};
  const float* agg_be[4] = {(const float*)d_in[22], (const float*)d_in[23],
                            (const float*)d_in[24], (const float*)d_in[25]};
  const float* pred_w = (const float*)d_in[26];
  const float* pred_b = (const float*)d_in[27];

  const int M = BATCH * TS;      // 16256 = 127 * 128
  char* wsp = (char*)d_ws;
  auto alloc = [&](size_t bytes) -> char* {
    char* p = wsp; wsp += (bytes + 255) & ~(size_t)255; return p;
  };
  float*    h1    = (float*)alloc((size_t)BATCH * T1 * 32 * 4);
  float*    h2    = (float*)alloc((size_t)BATCH * TS * 64 * 4);
  float*    z     = (float*)alloc((size_t)M * ZD * 4);
  float*    zha   = (float*)alloc((size_t)M * ZD * 4);
  float*    zhb   = (float*)alloc((size_t)M * ZD * 4);
  float*    ctmp  = (float*)alloc((size_t)M * ZD * 4);
  float*    zp    = (float*)alloc((size_t)M * ZD * 4);
  _Float16* w3h   = (_Float16*)alloc((size_t)64 * 128 * 2);
  _Float16* w4h   = (_Float16*)alloc((size_t)128 * 256 * 2);
  _Float16* aggh  = (_Float16*)alloc((size_t)14 * ZD * ZD * 2);
  _Float16* predh = (_Float16*)alloc((size_t)STEPS * ZD * ZD * 2);

  // ---- encoder front (VALU) ----
  {
    int n1 = BATCH * T1 * 32;
    conv1_kernel<<<(n1 + 255) / 256, 256, 0, stream>>>(x, enc_w[0], enc_b[0], h1);
    int n2 = BATCH * TS * 64;
    conv2_kernel<<<(n2 + 255) / 256, 256, 0, stream>>>(h1, enc_w[1], enc_b[1], h2);
  }
  // ---- weight prep: f32 -> f16 WMMA fragment packs ----
  wprep_kernel<<<(128 * 64 + 255) / 256, 256, 0, stream>>>(enc_w[2], w3h, 128, 64, 1);
  wprep_kernel<<<(256 * 128 + 255) / 256, 256, 0, stream>>>(enc_w[3], w4h, 256, 128, 1);
  const int ks[4] = {2, 3, 4, 5};
  const int aggoff[4] = {0, 2, 5, 9};
  for (int i = 0; i < 4; i++) {
    int tot = ZD * ZD * ks[i];
    wprep_kernel<<<(tot + 255) / 256, 256, 0, stream>>>(agg_w[i],
        aggh + (size_t)aggoff[i] * ZD * ZD, ZD, ZD, ks[i]);
  }
  predprep_kernel<<<(ZD * ZD * STEPS + 255) / 256, 256, 0, stream>>>(pred_w, predh);

  // ---- encoder back (WMMA GEMMs): conv3 (1x1, 64->128), conv4 (1x1, 128->256)
  gemm_wmma_kernel<<<dim3(1, M / 128), 256, 0, stream>>>(
      h2, w3h, enc_b[2], ctmp, 128, 64, TS, 1, 1);
  gemm_wmma_kernel<<<dim3(2, M / 128), 256, 0, stream>>>(
      ctmp, w4h, enc_b[3], z, 256, 128, TS, 1, 1);

  // ---- aggregator: 4 causal conv blocks (shifted-GEMM) + GN/relu/residual --
  const float* cur = z;
  float* bufs[2] = {zha, zhb};
  for (int i = 0; i < 4; i++) {
    gemm_wmma_kernel<<<dim3(2, M / 128), 256, 0, stream>>>(
        cur, aggh + (size_t)aggoff[i] * ZD * ZD, agg_b[i], ctmp,
        256, 256, TS, ks[i], 0);
    gn_res_kernel<<<BATCH, 256, 0, stream>>>(ctmp, cur, agg_g[i], agg_be[i], bufs[i & 1]);
    cur = bufs[i & 1];
  }

  // ---- predictor steps (WMMA GEMM) + CPC logits ----
  long long obase = 0;
  for (int i = 0; i < STEPS; i++) {
    gemm_wmma_kernel<<<dim3(2, M / 128), 256, 0, stream>>>(
        cur, predh + (size_t)i * ZD * ZD, pred_b, zp, 256, 256, TS, 1, 0);
    int Tlen = TS - OFF0 - i;
    int tot = Tlen * BATCH * 16;
    cpc_kernel<<<(tot + 255) / 256, 256, 0, stream>>>(
        zp, z, neg, (float*)d_out, i, Tlen, obase);
    obase += (long long)tot;
  }
  // ---- labels (int32 zeros == float 0.0 bit pattern) ----
  long long nz = (long long)out_size - obase;
  if (nz > 0)
    zerof_kernel<<<(int)((nz + 255) / 256), 256, 0, stream>>>(
        (float*)d_out, obase, (long long)out_size);
}
